// SparseMoETransformer_71356586655779
// MI455X (gfx1250) — matrix operand order repaired and checked by reading, not compiled, not run on hardware
//
#include <hip/hip_runtime.h>
#include <math.h>

typedef __bf16 bf16;
typedef __attribute__((ext_vector_type(16))) __bf16 v16bf;
typedef __attribute__((ext_vector_type(8)))  float  v8f;
typedef __attribute__((ext_vector_type(4)))  unsigned int v4u;
typedef __attribute__((ext_vector_type(8)))  int v8i;
typedef __attribute__((ext_vector_type(4)))  int v4i;

#if __has_builtin(__builtin_amdgcn_tensor_load_to_lds) && __has_builtin(__builtin_amdgcn_s_wait_tensorcnt)
#define USE_TDM 1
#else
#define USE_TDM 0
#endif

namespace {
constexpr int D   = 1024;   // d_model
constexpr int H   = 8;      // heads
constexpr int DH  = 128;    // head dim
constexpr int L   = 12;     // layers
constexpr int E   = 4;      // experts
constexpr int F   = 2048;   // ffn dim
constexpr int NC  = 10;     // classes
constexpr int BB  = 4;      // batch
constexpr int SS  = 512;    // seq
constexpr int NT  = BB * SS;      // 2048 tokens
constexpr int LSTR = 40;          // padded LDS row stride (bf16): 64B data + 16B pad
}

#if USE_TDM
// ---------------------------------------------------------------------------
// TDM: DMA one 2D tile (tile_dim0 = 32 bf16 = 64B per row, tile_dim1 = 128
// rows, row stride = ld elements) from global memory into LDS, inserting a
// 16B pad after every 64B so rows land at the 80B (LSTR) padded stride.
// D# layout per CDNA5 ISA ch.8: group0 = {count|flags, lds_addr, gaddr_lo,
// gaddr_hi|type=2}; group1 = {dsize|pad cfg, dims, tile dims, strides}.
// ---------------------------------------------------------------------------
__device__ inline void tdm_load_tile(unsigned lds_off, const bf16* gp, int ld)
{
    const unsigned long long ga = (unsigned long long)(uintptr_t)gp;
    v4u g0;
    g0[0] = 1u;                                    // count=1 (valid), user mode
    g0[1] = lds_off;                               // LDS byte address of tile
    g0[2] = (unsigned)(ga & 0xffffffffull);        // global_addr[31:0]
    g0[3] = (unsigned)((ga >> 32) & 0x1ffffffull)  // global_addr[56:32]
          | (2u << 30);                            // type = 2 ("image")
    const unsigned d0 = 1u << 20;                  // tensor_dim0 (OOB bound, ample)
    const unsigned d1 = 1u << 20;                  // tensor_dim1
    v8i g1;
    g1[0] = (int)((1u << 16)                       // data_size = 2 bytes
                | (1u << 20)                       // pad_enable
                | (3u << 22)                       // pad_interval: 16 DWORDs (64B)
                | (3u << 25));                     // pad_amount:   4 DWORDs (16B)
    g1[1] = (int)((d0 & 0xffffu) << 16);           // [31:16] = tensor_dim0 lo
    g1[2] = (int)((d0 >> 16) | ((d1 & 0xffffu) << 16));
    g1[3] = (int)((d1 >> 16) | (32u << 16));       // tile_dim0 = 32 elements
    g1[4] = 128;                                   // tile_dim1 = 128 rows, tile_dim2 = 0
    g1[5] = (int)(unsigned)ld;                     // tensor_dim0_stride[31:0]
    g1[6] = 0;                                     // stride hi / dim1_stride lo
    g1[7] = 0;
    const v4i zz = {0, 0, 0, 0};                   // 2D tensor: groups 2/3 unused
#if defined(__clang_major__) && (__clang_major__ >= 23)
    const v8i z8 = {0, 0, 0, 0, 0, 0, 0, 0};
    __builtin_amdgcn_tensor_load_to_lds(g0, g1, zz, zz, z8, 0);
#else
    __builtin_amdgcn_tensor_load_to_lds(g0, g1, zz, zz, 0);
#endif
}
#endif

// ---------------------------------------------------------------------------
// Generic batched GEMM:  C[z] = act( alpha * A[z] @ op(B[z]) + bias[z] )
//   A:  [M,K] bf16 row-major, lda
//   B:  BTRANS=1 -> weight [N,K] row-major (x @ W^T), ldb = K-stride
//       BTRANS=0 -> [K,N] row-major (attn @ v),       ldb = N-stride
//   C:  [M,N], ldc; written as f32 (outF) and/or bf16 (outB)
//   per-z offsets: off = (z/zdiv)*s1 + (z%zdiv)*s2
// Block: 256 threads = 8 wave32 waves; 128x128 tile; K step 32.
// Tiles are staged to LDS by the Tensor Data Mover (wave 0 issues two
// tensor_load_to_lds + s_wait_tensorcnt); each wave owns a 16-row strip,
// preloads 8 B fragments and issues 8 v_wmma_f32_16x16x32_bf16 per K-step.
// ---------------------------------------------------------------------------
template<int BTRANS, int ACT>
__global__ __launch_bounds__(256)
void gemm_bf16_wmma(const bf16* __restrict__ A, const bf16* __restrict__ Bm,
                    const float* __restrict__ bias,
                    float* __restrict__ outF, bf16* __restrict__ outB,
                    int M, int N, int K, int lda, int ldb, int ldc,
                    float alpha, int zdiv,
                    long long sA1, long long sA2,
                    long long sB1, long long sB2,
                    long long sC1, long long sC2,
                    long long sBias1, long long sBias2)
{
    __shared__ bf16 lA[128 * LSTR];
    __shared__ bf16 lB[128 * LSTR];

    const int tid  = threadIdx.x;
    const int wave = tid >> 5;          // 0..7
    const int lane = tid & 31;
    const int m0 = blockIdx.y * 128;
    const int n0 = blockIdx.x * 128;

    const long long z  = blockIdx.z;
    const long long zq = z / zdiv, zr = z % zdiv;
    const long long oA = zq * sA1 + zr * sA2;
    const long long oB = zq * sB1 + zr * sB2;
    const long long oC = zq * sC1 + zr * sC2;
    const long long oBias = zq * sBias1 + zr * sBias2;

    const v8f vzero = {0.f,0.f,0.f,0.f,0.f,0.f,0.f,0.f};
    v8f acc[8];
#pragma unroll
    for (int t = 0; t < 8; ++t) acc[t] = vzero;

    const int arow  = tid >> 1;        // 0..127
    const int ahalf = (tid & 1) * 16;  // 0 or 16 (K half)
#if USE_TDM
    const unsigned ldsA_off = (unsigned)(uintptr_t)&lA[0];
    const unsigned ldsB_off = (unsigned)(uintptr_t)&lB[0];
#endif

    for (int k0 = 0; k0 < K; k0 += 32) {
#if USE_TDM
        // ---- TDM-staged tiles (pad_enable reproduces the 80B LDS stride) ----
        if (BTRANS) {
            if (wave == 0) {
                tdm_load_tile(ldsA_off, A  + oA + (long long)m0 * lda + k0, lda);
                tdm_load_tile(ldsB_off, Bm + oB + (long long)n0 * ldb + k0, ldb);
                __builtin_amdgcn_s_wait_tensorcnt((short)0);
            }
        } else {
            if (wave == 0)
                tdm_load_tile(ldsA_off, A + oA + (long long)m0 * lda + k0, lda);
            const int bn = tid & 127;
            const int bk = (tid >> 7) * 16;
            const bf16* srcB = Bm + oB + (long long)(k0 + bk) * ldb + n0 + bn;
            bf16* dst = &lB[bn * LSTR + bk];
#pragma unroll
            for (int k = 0; k < 16; ++k) dst[k] = srcB[(long long)k * ldb];
            if (wave == 0) __builtin_amdgcn_s_wait_tensorcnt((short)0);
        }
#else
        // ---- manual staging fallback ----
        const bf16* srcA = A + oA + (long long)(m0 + arow) * lda + k0 + ahalf;
        {
            uint4* dst = (uint4*)&lA[arow * LSTR + ahalf];
            dst[0] = ((const uint4*)srcA)[0];
            dst[1] = ((const uint4*)srcA)[1];
        }
        if (BTRANS) {
            const bf16* srcB = Bm + oB + (long long)(n0 + arow) * ldb + k0 + ahalf;
            uint4* dst = (uint4*)&lB[arow * LSTR + ahalf];
            dst[0] = ((const uint4*)srcB)[0];
            dst[1] = ((const uint4*)srcB)[1];
            if (k0 + 32 < K) {
                __builtin_prefetch((const void*)(srcA + 32), 0, 3);
                __builtin_prefetch((const void*)(srcB + 32), 0, 3);
            }
        } else {
            const int bn = tid & 127;
            const int bk = (tid >> 7) * 16;
            const bf16* srcB = Bm + oB + (long long)(k0 + bk) * ldb + n0 + bn;
            bf16* dst = &lB[bn * LSTR + bk];
#pragma unroll
            for (int k = 0; k < 16; ++k) dst[k] = srcB[(long long)k * ldb];
            if (k0 + 32 < K) __builtin_prefetch((const void*)(srcA + 32), 0, 3);
        }
#endif
        __syncthreads();

        // ---- A fragment (CDNA5 16x32 bf16 layout) ----
        union Frag { v16bf v; uint4 q[2]; };
        Frag fa;
        {
            const int r  = (wave << 4) + (lane & 15);
            const int kb = (lane < 16) ? 0 : 8;
            const bf16* p = &lA[r * LSTR + kb];
            fa.q[0] = *(const uint4*)(p);        // K kb..kb+7
            fa.q[1] = *(const uint4*)(p + 16);   // K kb+16..kb+23
        }
        // ---- preload all 8 B fragments, then 8 WMMAs reusing fa ----
        Frag fb[8];
#pragma unroll
        for (int t = 0; t < 8; ++t) {
            const int c  = (t << 4) + (lane & 15);
            const int kb = (lane < 16) ? 0 : 8;
            const bf16* p = &lB[c * LSTR + kb];
            fb[t].q[0] = *(const uint4*)(p);     // K kb..kb+7
            fb[t].q[1] = *(const uint4*)(p + 16); // K kb+16..kb+23
        }
#pragma unroll
        for (int t = 0; t < 8; ++t)
            acc[t] = __builtin_amdgcn_wmma_f32_16x16x32_bf16(
                false, fa.v, false, fb[t].v, (short)0, acc[t], false, false);
        __syncthreads();
    }

    // ---- epilogue: alpha, bias, activation, dual-precision store ----
    const int nlo = lane & 15;
    const int mb  = m0 + (wave << 4) + ((lane < 16) ? 0 : 8);
#pragma unroll
    for (int t = 0; t < 8; ++t) {
        const int n = n0 + (t << 4) + nlo;
        float bv = bias ? bias[oBias + n] : 0.f;
#pragma unroll
        for (int i = 0; i < 8; ++i) {
            float v = acc[t][i] * alpha + bv;
            if (ACT == 1) v = 0.5f * v * (1.f + erff(v * 0.70710678118654752f)); // exact GELU
            const long long off = oC + (long long)(mb + i) * ldc + n;
            if (outF) outF[off] = v;
            if (outB) outB[off] = (bf16)v;
        }
    }
}

// ---------------------------------------------------------------------------
// Small fused kernels
// ---------------------------------------------------------------------------
__global__ void k_f32_to_bf16(const float* __restrict__ in, bf16* __restrict__ out,
                              long long n) {
    long long i = (long long)blockIdx.x * blockDim.x + threadIdx.x;
    const long long stride = (long long)gridDim.x * blockDim.x;
    for (; i < n; i += stride) out[i] = (bf16)in[i];
}

__global__ void k_add_pos(const float* __restrict__ x, const float* __restrict__ pos,
                          float* __restrict__ xf, bf16* __restrict__ xb) {
    const int i = blockIdx.x * blockDim.x + threadIdx.x;   // < NT*D
    const float v = x[i] + pos[i % (SS * D)];
    xf[i] = v; xb[i] = (bf16)v;
}

__global__ __launch_bounds__(256)
void k_ln_residual(const float* __restrict__ xin, const float* __restrict__ res,
                   const float* __restrict__ gam, const float* __restrict__ bet,
                   float* __restrict__ xf, bf16* __restrict__ xb) {
    __shared__ float sh[256];
    const int row = blockIdx.x, t = threadIdx.x;
    const long long base = (long long)row * D;
    float v[4]; float s = 0.f;
#pragma unroll
    for (int i = 0; i < 4; ++i) {
        const int d = t + i * 256;
        v[i] = xin[base + d] + res[base + d];
        s += v[i];
    }
    sh[t] = s; __syncthreads();
    for (int o = 128; o > 0; o >>= 1) { if (t < o) sh[t] += sh[t + o]; __syncthreads(); }
    const float mean = sh[0] * (1.f / D);
    __syncthreads();
    float s2 = 0.f;
#pragma unroll
    for (int i = 0; i < 4; ++i) { const float c = v[i] - mean; s2 += c * c; }
    sh[t] = s2; __syncthreads();
    for (int o = 128; o > 0; o >>= 1) { if (t < o) sh[t] += sh[t + o]; __syncthreads(); }
    const float rstd = rsqrtf(sh[0] * (1.f / D) + 1e-5f);
#pragma unroll
    for (int i = 0; i < 4; ++i) {
        const int d = t + i * 256;
        const float o = (v[i] - mean) * rstd * gam[d] + bet[d];
        xf[base + d] = o; xb[base + d] = (bf16)o;
    }
}

__global__ __launch_bounds__(256)
void k_softmax_row512(const float* __restrict__ sc, bf16* __restrict__ attn) {
    __shared__ float sh[256];
    const long long row = blockIdx.x;
    const int t = threadIdx.x;
    const float* p = sc + row * SS;
    const float a = p[t], b = p[t + 256];
    sh[t] = fmaxf(a, b); __syncthreads();
    for (int o = 128; o > 0; o >>= 1) { if (t < o) sh[t] = fmaxf(sh[t], sh[t + o]); __syncthreads(); }
    const float mx = sh[0]; __syncthreads();
    const float ea = __expf(a - mx), eb = __expf(b - mx);
    sh[t] = ea + eb; __syncthreads();
    for (int o = 128; o > 0; o >>= 1) { if (t < o) sh[t] += sh[t + o]; __syncthreads(); }
    const float inv = 1.f / sh[0];
    bf16* q = attn + row * SS;
    q[t] = (bf16)(ea * inv); q[t + 256] = (bf16)(eb * inv);
}

__global__ __launch_bounds__(256)
void k_gate_combine(const float* __restrict__ Wg, const float* __restrict__ bg,
                    const float* __restrict__ eo,
                    float* __restrict__ xf, bf16* __restrict__ xb) {
    __shared__ float sh[256];
    __shared__ float gl[E];
    __shared__ int   si[2];
    __shared__ float sv[2];
    const int row = blockIdx.x, t = threadIdx.x;
    const long long base = (long long)row * D;
    float p[E] = {0.f, 0.f, 0.f, 0.f};
    for (int d = t; d < D; d += 256) {
        const float xv = xf[base + d];
#pragma unroll
        for (int e = 0; e < E; ++e) p[e] += xv * Wg[e * D + d];
    }
    for (int e = 0; e < E; ++e) {
        sh[t] = p[e]; __syncthreads();
        for (int o = 128; o > 0; o >>= 1) { if (t < o) sh[t] += sh[t + o]; __syncthreads(); }
        if (t == 0) gl[e] = sh[0] + bg[e];
        __syncthreads();
    }
    if (t == 0) {  // softmax over 4 + top-2 (lower index wins ties, like lax.top_k)
        float mx = gl[0];
        for (int e = 1; e < E; ++e) mx = fmaxf(mx, gl[e]);
        float ex[E], sum = 0.f;
        for (int e = 0; e < E; ++e) { ex[e] = __expf(gl[e] - mx); sum += ex[e]; }
        int i0 = 0; for (int e = 1; e < E; ++e) if (ex[e] > ex[i0]) i0 = e;
        int i1 = (i0 == 0) ? 1 : 0;
        for (int e = 0; e < E; ++e) if (e != i0 && ex[e] > ex[i1]) i1 = e;
        const float inv = 1.f / sum;
        si[0] = i0; si[1] = i1; sv[0] = ex[i0] * inv; sv[1] = ex[i1] * inv;
    }
    __syncthreads();
    const long long MD = (long long)NT * D;
    for (int d = t; d < D; d += 256) {
        const float xv = xf[base + d];
        const float ev = sv[0] * eo[(long long)si[0] * MD + base + d]
                       + sv[1] * eo[(long long)si[1] * MD + base + d];
        const float nv = 0.5f * xv + 0.5f * ev;   // SPARSITY = 0.5
        xf[base + d] = nv; xb[base + d] = (bf16)nv;
    }
}

__global__ __launch_bounds__(256)
void k_classifier(const float* __restrict__ xf, const float* __restrict__ Wf,
                  const float* __restrict__ bfv, float* __restrict__ out) {
    __shared__ float sh[256];
    const int b = blockIdx.x, t = threadIdx.x;
    const long long base = (long long)(b * SS + SS - 1) * D;  // x[:, -1, :]
    for (int c = 0; c < NC; ++c) {
        float s = 0.f;
        for (int d = t; d < D; d += 256) s += xf[base + d] * Wf[(long long)c * D + d];
        sh[t] = s; __syncthreads();
        for (int o = 128; o > 0; o >>= 1) { if (t < o) sh[t] += sh[t + o]; __syncthreads(); }
        if (t == 0) out[b * NC + c] = sh[0] + bfv[c];
        __syncthreads();
    }
}

// ---------------------------------------------------------------------------
// Orchestration
// ---------------------------------------------------------------------------
extern "C" void kernel_launch(void* const* d_in, const int* in_sizes, int n_in,
                              void* d_out, int out_size, void* d_ws, size_t ws_size,
                              hipStream_t stream) {
    (void)in_sizes; (void)n_in; (void)out_size; (void)ws_size;
    const float* x     = (const float*)d_in[0];
    const float* pos   = (const float*)d_in[1];
    const float* Wqkv  = (const float*)d_in[2];
    const float* bqkv  = (const float*)d_in[3];
    const float* Wo    = (const float*)d_in[4];
    const float* bo    = (const float*)d_in[5];
    const float* ln1s  = (const float*)d_in[6];
    const float* ln1b  = (const float*)d_in[7];
    const float* W1    = (const float*)d_in[8];
    const float* b1    = (const float*)d_in[9];
    const float* W2    = (const float*)d_in[10];
    const float* b2    = (const float*)d_in[11];
    const float* ln2s  = (const float*)d_in[12];
    const float* ln2b  = (const float*)d_in[13];
    const float* Wg    = (const float*)d_in[14];
    const float* bg    = (const float*)d_in[15];
    const float* We    = (const float*)d_in[16];
    const float* be    = (const float*)d_in[17];
    const float* Wf    = (const float*)d_in[18];
    const float* bf_   = (const float*)d_in[19];
    float* out = (float*)d_out;

    // workspace partition
    char* w = (char*)d_ws;
    auto alloc = [&](size_t bytes) -> void* {
        void* p = (void*)w; w += (bytes + 255) & ~(size_t)255; return p;
    };
    const size_t nWqkv = (size_t)L * 3 * D * D;
    const size_t nWo   = (size_t)L * D * D;
    const size_t nW1   = (size_t)L * F * D;
    const size_t nW2   = (size_t)L * D * F;
    const size_t nWe   = (size_t)E * D * D;
    bf16* wq_b = (bf16*)alloc(nWqkv * 2);
    bf16* wo_b = (bf16*)alloc(nWo * 2);
    bf16* w1_b = (bf16*)alloc(nW1 * 2);
    bf16* w2_b = (bf16*)alloc(nW2 * 2);
    bf16* we_b = (bf16*)alloc(nWe * 2);
    float* xf   = (float*)alloc((size_t)NT * D * 4);
    bf16*  xb   = (bf16*) alloc((size_t)NT * D * 2);
    bf16*  qkvb = (bf16*) alloc((size_t)NT * 3 * D * 2);
    float* scf  = (float*)alloc((size_t)BB * H * SS * SS * 4);
    bf16*  attb = (bf16*) alloc((size_t)BB * H * SS * SS * 2);
    bf16*  ob   = (bf16*) alloc((size_t)NT * D * 2);
    float* of   = (float*)alloc((size_t)NT * D * 4);
    bf16*  h1b  = (bf16*) alloc((size_t)NT * F * 2);
    float* h2f  = (float*)alloc((size_t)NT * D * 4);
    float* eo   = (float*)alloc((size_t)E * NT * D * 4);

    // weight conversion fp32 -> bf16 (every call: launch must be stateless)
    k_f32_to_bf16<<<8192, 256, 0, stream>>>(Wqkv, wq_b, (long long)nWqkv);
    k_f32_to_bf16<<<8192, 256, 0, stream>>>(Wo,   wo_b, (long long)nWo);
    k_f32_to_bf16<<<8192, 256, 0, stream>>>(W1,   w1_b, (long long)nW1);
    k_f32_to_bf16<<<8192, 256, 0, stream>>>(W2,   w2_b, (long long)nW2);
    k_f32_to_bf16<<<8192, 256, 0, stream>>>(We,   we_b, (long long)nWe);

    // x = x + pos
    k_add_pos<<<(NT * D) / 256, 256, 0, stream>>>(x, pos, xf, xb);

    const long long sQ1 = (long long)SS * 3 * D;   // per-batch stride in qkv buffer
    const long long sQ2 = DH;                      // per-head stride
    const long long sS1 = (long long)H * SS * SS;  // per-batch stride in scores
    const long long sS2 = (long long)SS * SS;      // per-head stride
    const float rsdh = 0.08838834764831845f;       // 1/sqrt(128)

    for (int l = 0; l < L; ++l) {
        // ---- qkv = x @ Wqkv^T + bqkv  -> bf16 [NT, 3D] ----
        gemm_bf16_wmma<1, 0><<<dim3((3 * D) / 128, NT / 128, 1), 256, 0, stream>>>(
            xb, wq_b + (size_t)l * 3 * D * D, bqkv + (size_t)l * 3 * D,
            nullptr, qkvb, NT, 3 * D, D, D, D, 3 * D,
            1.f, 1, 0, 0, 0, 0, 0, 0, 0, 0);
        // ---- scores = (q @ k^T) / sqrt(DH), batched over B*H ----
        gemm_bf16_wmma<1, 0><<<dim3(SS / 128, SS / 128, BB * H), 256, 0, stream>>>(
            qkvb /*q*/, qkvb + D /*k*/, nullptr, scf, nullptr,
            SS, SS, DH, 3 * D, 3 * D, SS,
            rsdh, H, sQ1, sQ2, sQ1, sQ2, sS1, sS2, 0, 0);
        // ---- softmax rows -> bf16 attention probs ----
        k_softmax_row512<<<BB * H * SS, 256, 0, stream>>>(scf, attb);
        // ---- o = attn @ v, scattered to [B,S,D] head slices ----
        gemm_bf16_wmma<0, 0><<<dim3(DH / 128, SS / 128, BB * H), 256, 0, stream>>>(
            attb, qkvb + 2 * D /*v*/, nullptr, nullptr, ob,
            SS, DH, SS, SS, 3 * D, D,
            1.f, H, sS1, sS2, sQ1, sQ2, (long long)SS * D, DH, 0, 0);
        // ---- o @ Wo^T + bo ----
        gemm_bf16_wmma<1, 0><<<dim3(D / 128, NT / 128, 1), 256, 0, stream>>>(
            ob, wo_b + (size_t)l * D * D, bo + (size_t)l * D,
            of, nullptr, NT, D, D, D, D, D,
            1.f, 1, 0, 0, 0, 0, 0, 0, 0, 0);
        // ---- x = LN(x + o) ----
        k_ln_residual<<<NT, 256, 0, stream>>>(xf, of, ln1s + (size_t)l * D,
                                              ln1b + (size_t)l * D, xf, xb);
        // ---- h1 = gelu(x @ W1^T + b1) -> bf16 ----
        gemm_bf16_wmma<1, 1><<<dim3(F / 128, NT / 128, 1), 256, 0, stream>>>(
            xb, w1_b + (size_t)l * F * D, b1 + (size_t)l * F,
            nullptr, h1b, NT, F, D, D, D, F,
            1.f, 1, 0, 0, 0, 0, 0, 0, 0, 0);
        // ---- h2 = h1 @ W2^T + b2 ----
        gemm_bf16_wmma<1, 0><<<dim3(D / 128, NT / 128, 1), 256, 0, stream>>>(
            h1b, w2_b + (size_t)l * D * F, b2 + (size_t)l * D,
            h2f, nullptr, NT, D, F, F, F, D,
            1.f, 1, 0, 0, 0, 0, 0, 0, 0, 0);
        // ---- x = LN(x + h2) ----
        k_ln_residual<<<NT, 256, 0, stream>>>(xf, h2f, ln2s + (size_t)l * D,
                                              ln2b + (size_t)l * D, xf, xb);
        // ---- all 4 expert outputs (dense, matching reference) ----
        gemm_bf16_wmma<1, 0><<<dim3(D / 128, NT / 128, E), 256, 0, stream>>>(
            xb, we_b, be, eo, nullptr, NT, D, D, D, D, D,
            1.f, 1, 0, 0, (long long)D * D, 0, (long long)NT * D, 0,
            (long long)D, 0);
        // ---- gate softmax + top-2 + 0.5/0.5 mix ----
        k_gate_combine<<<NT, 256, 0, stream>>>(Wg, bg, eo, xf, xb);
    }

    // ---- classifier on last token ----
    k_classifier<<<BB, 256, 0, stream>>>(xf, Wf, bf_, out);
}
